// VPTCausalPolicy_19439021981919
// MI455X (gfx1250) — matrix operand order repaired and checked by reading, not compiled, not run on hardware
//
#include <hip/hip_runtime.h>

constexpr int L_ = 6, D_ = 768, H_ = 8, DH_ = 96, FF_ = 2048, A_ = 4;
constexpr int B_ = 8, T_ = 1024, TOK_ = B_ * T_;
constexpr float EPS_ = 1e-5f;
constexpr float SCALE_ = 0.10206207261596577f; // 1/sqrt(96)

typedef __attribute__((ext_vector_type(16))) __bf16 v16bf;
typedef __attribute__((ext_vector_type(8)))  float  v8f;

union BFrag { unsigned int u[8]; v16bf v; };

__device__ __forceinline__ unsigned short f2bf(float f) {
  unsigned int u = __float_as_uint(f);
  u += 0x7FFFu + ((u >> 16) & 1u);            // round-to-nearest-even
  return (unsigned short)(u >> 16);
}

__device__ __forceinline__ v8f wmma_bf(BFrag a, BFrag b, v8f c) {
  return __builtin_amdgcn_wmma_f32_16x16x32_bf16(false, a.v, false, b.v,
                                                 (short)0, c, false, false);
}

__device__ __forceinline__ float gelu_exact(float x) {
  return 0.5f * x * (1.0f + erff(x * 0.7071067811865476f));
}

__device__ __forceinline__ int clampi(int v, int lo, int hi) {
  return v < lo ? lo : (v > hi ? hi : v);
}

// ---------------------------------------------------------------- fp32->bf16
__global__ __launch_bounds__(256) void f2bf_kernel(const float* __restrict__ in,
                                                   unsigned short* __restrict__ out,
                                                   long long n) {
  long long i = (long long)blockIdx.x * 256 + threadIdx.x;
  if (i < n) out[i] = f2bf(in[i]);
}

// ------------------------------------- fp32->bf16 with per-matrix transpose
// in: mats x [K, N] row-major; out: mats x [N, K] row-major (bf16)
__global__ __launch_bounds__(256) void f2bf_t_kernel(const float* __restrict__ in,
                                                     unsigned short* __restrict__ out,
                                                     int mats, int K, int N) {
  long long i = (long long)blockIdx.x * 256 + threadIdx.x;
  long long tot = (long long)mats * K * N;
  if (i >= tot) return;
  int kn = K * N;
  int l = (int)(i / kn);
  int rem = (int)(i - (long long)l * kn);
  int n = rem / K;
  int k = rem - n * K;
  out[i] = f2bf(in[(long long)l * kn + (long long)k * N + n]);
}

// ------------------------------------------------------------- pos embedding
__global__ __launch_bounds__(256) void pos_embed_kernel(unsigned short* __restrict__ pos) {
  int i = blockIdx.x * 256 + threadIdx.x;
  if (i >= T_ * D_) return;
  int j = i / D_, d = i % D_;
  int f = (d < D_ / 2) ? d : d - D_ / 2;
  float inv = expf(-(float)(2 * f) * (9.210340371976184f / (float)D_)); // 10000^(-2f/D)
  float ang = (float)(T_ - 1 - j) * inv;
  float val = (d < D_ / 2) ? sinf(ang) : cosf(ang);
  pos[i] = f2bf(val);
}

// ------------------------------------------------------------------ layernorm
__global__ __launch_bounds__(256) void layernorm_kernel(
    const float* __restrict__ x, const float* __restrict__ g,
    const float* __restrict__ b, float* __restrict__ outF,
    unsigned short* __restrict__ outB) {
  __shared__ float red[16];
  const int t = threadIdx.x;
  const size_t row = blockIdx.x;
  const float* xr = x + row * D_;
  float vv[3];
  vv[0] = xr[t]; vv[1] = xr[t + 256]; vv[2] = xr[t + 512];
  float s = vv[0] + vv[1] + vv[2];
  float q = vv[0] * vv[0] + vv[1] * vv[1] + vv[2] * vv[2];
#pragma unroll
  for (int m = 16; m >= 1; m >>= 1) {
    s += __shfl_xor(s, m, 32);
    q += __shfl_xor(q, m, 32);
  }
  if ((t & 31) == 0) { red[t >> 5] = s; red[(t >> 5) + 8] = q; }
  __syncthreads();
  float ssum = 0.f, qsum = 0.f;
#pragma unroll
  for (int i = 0; i < 8; ++i) { ssum += red[i]; qsum += red[i + 8]; }
  float mean = ssum * (1.0f / D_);
  float var = qsum * (1.0f / D_) - mean * mean;
  float rstd = rsqrtf(var + EPS_);
#pragma unroll
  for (int e = 0; e < 3; ++e) {
    int i = t + e * 256;
    float y = (vv[e] - mean) * rstd * g[i] + b[i];
    if (outF) outF[row * D_ + i] = y;
    if (outB) outB[row * D_ + i] = f2bf(y);
  }
}

// ----------------------------------------------------------- bf16 WMMA GEMM
// C[M,N] = A[M,K] @ Bt[N,K]^T  (+bias) (+gelu) (+residual)
// 128x128 macro-tile, 8 waves x (32x64), K=32 steps, double-buffered LDS
// staged with GLOBAL_LOAD_ASYNC_TO_LDS_B128 (ASYNCcnt).
__global__ __launch_bounds__(256) void gemm_bf16_kernel(
    const unsigned short* __restrict__ A, const unsigned short* __restrict__ Bt,
    int M, int N, int K,
    const float* __restrict__ bias, const float* __restrict__ bias2,
    const float* __restrict__ resid,
    float* __restrict__ outF, unsigned short* __restrict__ outB,
    unsigned short* __restrict__ outB2, int do_gelu) {
  __shared__ alignas(16) unsigned short As[2][128 * 32];
  __shared__ alignas(16) unsigned short Bs[2][128 * 32];
  const int tid = threadIdx.x;
  const int lane = tid & 31;
  const int wv = tid >> 5;
  const int m0 = blockIdx.y * 128;
  const int n0 = blockIdx.x * 128;
  const int mwave = wv >> 1;   // 0..3 -> 32 rows each
  const int nwave = wv & 1;    // 0..1 -> 64 cols each
  const int nn = lane & 15;
  const int grp = lane >> 4;

  const unsigned long long abase = (unsigned long long)A;
  const unsigned long long bbase = (unsigned long long)Bt;

  // issue one 128x32 A tile + 128x32 B tile (both k-contiguous) as async copies
  auto issue_tile = [&](int kk, int buf) {
#pragma unroll
    for (int e = 0; e < 2; ++e) {
      int idx = tid + e * 256;        // 0..511 b128 chunks per tile
      int row = idx >> 1;             // 0..255
      int col = (idx & 1) * 8;        // 0 or 8 (bf16 elems)
      unsigned ldsA = (unsigned)(size_t)(const void*)&As[buf][row * 32 + col];
      unsigned offA = (unsigned)(((unsigned)(m0 + row) * (unsigned)K + kk + col) * 2u);
      asm volatile("global_load_async_to_lds_b128 %0, %1, %2 offset:0"
                   :: "v"(ldsA), "v"(offA), "s"(abase) : "memory");
      unsigned ldsB = (unsigned)(size_t)(const void*)&Bs[buf][row * 32 + col];
      unsigned offB = (unsigned)(((unsigned)(n0 + row) * (unsigned)K + kk + col) * 2u);
      asm volatile("global_load_async_to_lds_b128 %0, %1, %2 offset:0"
                   :: "v"(ldsB), "v"(offB), "s"(bbase) : "memory");
    }
  };

  v8f acc[2][4] = {};

  issue_tile(0, 0);
  const int nk = K / 32;
  for (int i = 0; i < nk; ++i) {
    asm volatile("s_wait_asynccnt 0x0" ::: "memory");
    __syncthreads();
    if (i + 1 < nk) issue_tile((i + 1) * 32, (i + 1) & 1);
    const unsigned short* as_ = &As[i & 1][0];
    const unsigned short* bs_ = &Bs[i & 1][0];
    BFrag fa[2], fb[4];
#pragma unroll
    for (int mt = 0; mt < 2; ++mt) {
#pragma unroll
      for (int v = 0; v < 8; ++v) {
        int ka = grp * 16 + 2 * v;                        // 16-bit A layout (contiguous k)
        fa[mt].u[v] = *(const unsigned int*)&as_[(mwave * 32 + mt * 16 + nn) * 32 + ka];
      }
    }
#pragma unroll
    for (int nt = 0; nt < 4; ++nt) {
#pragma unroll
      for (int v = 0; v < 8; ++v) {
        int kb = grp * 16 + 2 * v;                        // 16-bit B layout
        fb[nt].u[v] = *(const unsigned int*)&bs_[(nwave * 64 + nt * 16 + nn) * 32 + kb];
      }
    }
#pragma unroll
    for (int mt = 0; mt < 2; ++mt)
#pragma unroll
      for (int nt = 0; nt < 4; ++nt)
        acc[mt][nt] = wmma_bf(fa[mt], fb[nt], acc[mt][nt]);
  }

  const int mb = grp * 8;
#pragma unroll
  for (int mt = 0; mt < 2; ++mt) {
#pragma unroll
    for (int nt = 0; nt < 4; ++nt) {
      int gn = n0 + nwave * 64 + nt * 16 + nn;
#pragma unroll
      for (int r = 0; r < 8; ++r) {
        int gm = m0 + mwave * 32 + mt * 16 + mb + r;
        size_t idx = (size_t)gm * N + gn;
        float xraw = acc[mt][nt][r];
        float x1 = xraw;
        if (bias) x1 += bias[gn];
        if (do_gelu) x1 = gelu_exact(x1);
        if (resid) x1 += resid[idx];
        if (outF) outF[idx] = x1;
        if (outB) outB[idx] = f2bf(x1);
        if (outB2) outB2[idx] = f2bf(xraw + bias2[gn]);
      }
    }
  }
}

// -------------------------------------------- fused rel-pos flash attention
// One wave per (b, h, 16-row q tile). k streamed in steps of 32.
__global__ __launch_bounds__(256) void attn_kernel(
    const unsigned short* __restrict__ qw, const unsigned short* __restrict__ qr,
    const unsigned short* __restrict__ kx, const unsigned short* __restrict__ vx,
    const unsigned short* __restrict__ rx, unsigned short* __restrict__ av) {
  __shared__ alignas(16) float Gbuf[8][16 * 48];
  __shared__ alignas(16) unsigned short Pbuf[8][16 * 32];
  const int lane = threadIdx.x & 31;
  const int wv = threadIdx.x >> 5;
  const int gw = blockIdx.x * 8 + wv;
  const int qt = gw & (T_ / 16 - 1);
  const int bh = gw >> 6;
  const int hh = bh & (H_ - 1);
  const int bb = bh >> 3;
  const int q0 = qt * 16;
  const int nn = lane & 15;
  const int grp = lane >> 4;
  const int mb = grp * 8;

  // A-fragments of (q+rwb) and (q+rrb): 16 rows x 96 (3 chunks of K=32)
  BFrag aw[3], ar[3];
  {
    const size_t rowoff = (size_t)(bb * T_ + q0 + nn) * D_ + (size_t)hh * DH_;
#pragma unroll
    for (int c = 0; c < 3; ++c) {
#pragma unroll
      for (int v = 0; v < 8; ++v) {
        int ka = grp * 16 + 2 * v;
        aw[c].u[v] = *(const unsigned int*)(qw + rowoff + c * 32 + ka);
        ar[c].u[v] = *(const unsigned int*)(qr + rowoff + c * 32 + ka);
      }
    }
  }

  float mstat[8], sstat[8];
  v8f o[6] = {};
#pragma unroll
  for (int r = 0; r < 8; ++r) { mstat[r] = -3.0e38f; sstat[r] = 0.0f; }

  float* gb = &Gbuf[wv][0];
  unsigned short* pb = &Pbuf[wv][0];

  for (int k0 = 0; k0 <= q0 + 15; k0 += 32) {
    v8f s0 = {}, s1 = {}, g0 = {}, g1 = {}, g2 = {};
    const int gbase = k0 - q0 + T_ - 16;  // G col c -> R row gbase + c
#pragma unroll
    for (int c = 0; c < 3; ++c) {
      const size_t dcol = (size_t)hh * DH_ + c * 32 + grp * 16;
      const int r0 = clampi(gbase + nn, 0, T_ - 1);
      const int r1 = clampi(gbase + 16 + nn, 0, T_ - 1);
      const int r2 = clampi(gbase + 32 + nn, 0, T_ - 1);
      BFrag fk0, fk1, fr0, fr1, fr2;
#pragma unroll
      for (int v = 0; v < 8; ++v) {
        int dk = 2 * v;
        fk0.u[v] = *(const unsigned int*)(kx + (size_t)(bb * T_ + k0 + nn) * D_ + dcol + dk);
        fk1.u[v] = *(const unsigned int*)(kx + (size_t)(bb * T_ + k0 + 16 + nn) * D_ + dcol + dk);
        fr0.u[v] = *(const unsigned int*)(rx + (size_t)r0 * D_ + dcol + dk);
        fr1.u[v] = *(const unsigned int*)(rx + (size_t)r1 * D_ + dcol + dk);
        fr2.u[v] = *(const unsigned int*)(rx + (size_t)r2 * D_ + dcol + dk);
      }
      s0 = wmma_bf(aw[c], fk0, s0);
      s1 = wmma_bf(aw[c], fk1, s1);
      g0 = wmma_bf(ar[c], fr0, g0);
      g1 = wmma_bf(ar[c], fr1, g1);
      g2 = wmma_bf(ar[c], fr2, g2);
    }
    // spill G tile (16x48) to per-wave LDS for the rel-shift diagonal gather
#pragma unroll
    for (int r = 0; r < 8; ++r) {
      gb[(mb + r) * 48 + nn]      = g0[r];
      gb[(mb + r) * 48 + 16 + nn] = g1[r];
      gb[(mb + r) * 48 + 32 + nn] = g2[r];
    }
    asm volatile("s_wait_dscnt 0" ::: "memory");
    // online softmax update (row groups match C-layout lane grouping)
#pragma unroll
    for (int r = 0; r < 8; ++r) {
      int qi = mb + r;
      int qrow = q0 + qi;
      float sc0 = (k0 + nn <= qrow)
                      ? (s0[r] + gb[qi * 48 + (nn - qi + 15)]) * SCALE_ : -3.0e38f;
      float sc1 = (k0 + 16 + nn <= qrow)
                      ? (s1[r] + gb[qi * 48 + (nn - qi + 31)]) * SCALE_ : -3.0e38f;
      float t = fmaxf(sc0, sc1);
      t = fmaxf(t, __shfl_xor(t, 1, 32));
      t = fmaxf(t, __shfl_xor(t, 2, 32));
      t = fmaxf(t, __shfl_xor(t, 4, 32));
      t = fmaxf(t, __shfl_xor(t, 8, 32));
      float mnew = fmaxf(mstat[r], t);
      float alpha = __expf(mstat[r] - mnew);
      float e0 = __expf(sc0 - mnew);
      float e1 = __expf(sc1 - mnew);
      float su = e0 + e1;
      su += __shfl_xor(su, 1, 32);
      su += __shfl_xor(su, 2, 32);
      su += __shfl_xor(su, 4, 32);
      su += __shfl_xor(su, 8, 32);
      sstat[r] = sstat[r] * alpha + su;
      mstat[r] = mnew;
#pragma unroll
      for (int j = 0; j < 6; ++j) o[j][r] *= alpha;
      pb[qi * 32 + nn]      = f2bf(e0);
      pb[qi * 32 + 16 + nn] = f2bf(e1);
    }
    asm volatile("s_wait_dscnt 0" ::: "memory");
    // P as A-fragment (16x32)
    BFrag fp;
#pragma unroll
    for (int v = 0; v < 8; ++v) {
      int ka = grp * 16 + 2 * v;
      fp.u[v] = *(const unsigned int*)&pb[nn * 32 + ka];
    }
    // O += P @ V   (6 column tiles of 16 over DH=96)
#pragma unroll
    for (int j = 0; j < 6; ++j) {
      BFrag fv;
#pragma unroll
      for (int v = 0; v < 8; ++v) {
        int kk = grp * 16 + 2 * v;
        const unsigned short* vp =
            vx + (size_t)(bb * T_ + k0 + kk) * D_ + (size_t)hh * DH_ + j * 16 + nn;
        unsigned int lo = vp[0];
        unsigned int hi = vp[D_];
        fv.u[v] = lo | (hi << 16);
      }
      o[j] = wmma_bf(fp, fv, o[j]);
    }
  }
  // normalize and store av (bf16)
#pragma unroll
  for (int j = 0; j < 6; ++j) {
#pragma unroll
    for (int r = 0; r < 8; ++r) {
      float val = o[j][r] / sstat[r];
      av[(size_t)(bb * T_ + q0 + mb + r) * D_ + (size_t)hh * DH_ + j * 16 + nn] = f2bf(val);
    }
  }
}

// -------------------------------------------------------------- policy head
__global__ __launch_bounds__(256) void final_proj_kernel(
    const float* __restrict__ fin, const float* __restrict__ Wp,
    const float* __restrict__ bp, float* __restrict__ out) {
  int idx = blockIdx.x * 256 + threadIdx.x;
  if (idx >= TOK_ * A_) return;
  int a = idx & (A_ - 1);
  int bt = idx >> 2;
  const float* xr = fin + (size_t)bt * D_;
  float s = bp[a];
  for (int d = 0; d < D_; ++d) s += xr[d] * Wp[d * A_ + a];
  out[idx] = s;
}

// ------------------------------------------------------------------- driver
extern "C" void kernel_launch(void* const* d_in, const int* in_sizes, int n_in,
                              void* d_out, int out_size, void* d_ws, size_t ws_size,
                              hipStream_t stream) {
  (void)in_sizes; (void)n_in; (void)out_size; (void)ws_size;
  const float* x    = (const float*)d_in[0];
  const float* Wq   = (const float*)d_in[1];
  const float* Wk   = (const float*)d_in[2];
  const float* Wv   = (const float*)d_in[3];
  const float* Wr   = (const float*)d_in[4];
  const float* Wo   = (const float*)d_in[5];
  const float* rwb  = (const float*)d_in[6];
  const float* rrb  = (const float*)d_in[7];
  const float* ln1g = (const float*)d_in[8];
  const float* ln1b = (const float*)d_in[9];
  const float* ln2g = (const float*)d_in[10];
  const float* ln2b = (const float*)d_in[11];
  const float* W1   = (const float*)d_in[12];
  const float* b1   = (const float*)d_in[13];
  const float* W2   = (const float*)d_in[14];
  const float* b2   = (const float*)d_in[15];
  const float* ing  = (const float*)d_in[16];
  const float* inb  = (const float*)d_in[17];
  const float* fing = (const float*)d_in[18];
  const float* finb = (const float*)d_in[19];
  const float* Wp   = (const float*)d_in[20];
  const float* bp   = (const float*)d_in[21];
  float* out = (float*)d_out;

  char* base = (char*)d_ws;
  size_t off = 0;
  auto carve = [&](size_t bytes) -> void* {
    void* r = base + off;
    off += (bytes + 255) & ~(size_t)255;
    return r;
  };
  const size_t DD = (size_t)D_ * D_;
  const size_t DF = (size_t)D_ * FF_;
  unsigned short* bWq = (unsigned short*)carve(L_ * DD * 2);
  unsigned short* bWk = (unsigned short*)carve(L_ * DD * 2);
  unsigned short* bWv = (unsigned short*)carve(L_ * DD * 2);
  unsigned short* bWr = (unsigned short*)carve(L_ * DD * 2);
  unsigned short* bWo = (unsigned short*)carve(L_ * DD * 2);
  unsigned short* bW1 = (unsigned short*)carve(L_ * DF * 2);
  unsigned short* bW2 = (unsigned short*)carve(L_ * DF * 2);
  unsigned short* bPos = (unsigned short*)carve((size_t)T_ * D_ * 2);
  float* h = (float*)carve((size_t)TOK_ * D_ * 4);
  unsigned short* aBf  = (unsigned short*)carve((size_t)TOK_ * D_ * 2);
  unsigned short* qwBf = (unsigned short*)carve((size_t)TOK_ * D_ * 2);
  unsigned short* qrBf = (unsigned short*)carve((size_t)TOK_ * D_ * 2);
  unsigned short* kBf  = (unsigned short*)carve((size_t)TOK_ * D_ * 2);
  unsigned short* vBf  = (unsigned short*)carve((size_t)TOK_ * D_ * 2);
  unsigned short* avBf = (unsigned short*)carve((size_t)TOK_ * D_ * 2);
  unsigned short* rBf  = (unsigned short*)carve((size_t)T_ * D_ * 2);
  unsigned short* ffBf = (unsigned short*)carve((size_t)TOK_ * FF_ * 2);
  float* finF = (float*)carve((size_t)TOK_ * D_ * 4);

  // weights: convert + transpose to [N, K] bf16 so GEMM B tiles are k-contiguous
  auto cvtT = [&](const float* s, unsigned short* d, int mats, int K, int N) {
    long long n = (long long)mats * K * N;
    f2bf_t_kernel<<<dim3((unsigned)((n + 255) / 256)), dim3(256), 0, stream>>>(s, d, mats, K, N);
  };
  cvtT(Wq, bWq, L_, D_, D_);
  cvtT(Wk, bWk, L_, D_, D_);
  cvtT(Wv, bWv, L_, D_, D_);
  cvtT(Wr, bWr, L_, D_, D_);
  cvtT(Wo, bWo, L_, D_, D_);
  cvtT(W1, bW1, L_, D_, FF_);
  cvtT(W2, bW2, L_, FF_, D_);
  pos_embed_kernel<<<dim3((T_ * D_) / 256), dim3(256), 0, stream>>>(bPos);

  layernorm_kernel<<<dim3(TOK_), dim3(256), 0, stream>>>(x, ing, inb, h, nullptr);

  auto gemm = [&](const unsigned short* Am, const unsigned short* Bm, int M, int N, int K,
                  const float* bias, const float* bias2, const float* resid,
                  float* oF, unsigned short* oB, unsigned short* oB2, int gelu) {
    gemm_bf16_kernel<<<dim3(N / 128, M / 128), dim3(256), 0, stream>>>(
        Am, Bm, M, N, K, bias, bias2, resid, oF, oB, oB2, gelu);
  };

  for (int l = 0; l < L_; ++l) {
    layernorm_kernel<<<dim3(TOK_), dim3(256), 0, stream>>>(h, ln1g + l * D_, ln1b + l * D_,
                                                           nullptr, aBf);
    gemm(aBf, bWq + l * DD, TOK_, D_, D_, rwb + l * D_, rrb + l * D_, nullptr,
         nullptr, qwBf, qrBf, 0);
    gemm(aBf, bWk + l * DD, TOK_, D_, D_, nullptr, nullptr, nullptr, nullptr, kBf, nullptr, 0);
    gemm(aBf, bWv + l * DD, TOK_, D_, D_, nullptr, nullptr, nullptr, nullptr, vBf, nullptr, 0);
    gemm(bPos, bWr + l * DD, T_, D_, D_, nullptr, nullptr, nullptr, nullptr, rBf, nullptr, 0);
    attn_kernel<<<dim3((B_ * H_ * (T_ / 16)) / 8), dim3(256), 0, stream>>>(
        qwBf, qrBf, kBf, vBf, rBf, avBf);
    gemm(avBf, bWo + l * DD, TOK_, D_, D_, nullptr, nullptr, h, h, nullptr, nullptr, 0);
    layernorm_kernel<<<dim3(TOK_), dim3(256), 0, stream>>>(h, ln2g + l * D_, ln2b + l * D_,
                                                           nullptr, aBf);
    gemm(aBf, bW1 + l * DF, TOK_, FF_, D_, b1 + l * FF_, nullptr, nullptr,
         nullptr, ffBf, nullptr, 1);
    gemm(ffBf, bW2 + l * DF, TOK_, D_, FF_, b2 + l * D_, nullptr, h, h, nullptr, nullptr, 0);
  }
  layernorm_kernel<<<dim3(TOK_), dim3(256), 0, stream>>>(h, fing, finb, finF, nullptr);
  final_proj_kernel<<<dim3((TOK_ * A_) / 256), dim3(256), 0, stream>>>(finF, Wp, bp, out);
}
